// SlotAttentionModule_37787122270480
// MI455X (gfx1250) — matrix operand order repaired and checked by reading, not compiled, not run on hardware
//
#include <hip/hip_runtime.h>
#include <hip/hip_bf16.h>
#include <math.h>

typedef __bf16 bf16_t;
typedef __bf16  v16bf __attribute__((ext_vector_type(16)));
typedef __bf16  v8bf  __attribute__((ext_vector_type(8)));
typedef float   v8f   __attribute__((ext_vector_type(8)));

#define BATCH 64
#define NPTS  4096
#define DIN   256
#define KSLOT 20
#define KPAD  32
#define DSL   64
#define NT    256          // n-chunk per attn block
#define NCHUNK (NPTS / NT) // 16
#define SCALE 0.125f       // DS^-0.5
#define EPS_A 1e-8f
#define EPS_LN 1e-5f

// ---------------- WMMA helpers (gfx1250 wave32) ----------------

__device__ inline v8f wmma_bf16(v16bf a, v16bf b, v8f c) {
  return __builtin_amdgcn_wmma_f32_16x16x32_bf16(
      /*neg_a=*/false, a, /*neg_b=*/false, b,
      /*c_mod=*/(short)0, c, /*reuse_a=*/false, /*reuse_b=*/false);
}

// A fragment: 16x32 bf16, row-major source with row stride `ld` (elements).
// lane<16: row=lane, K = kk+[0..8) and kk+[16..24)
// lane>=16: row=lane-16, K = kk+[8..16) and kk+[24..32)
__device__ inline v16bf load_a_frag(const bf16_t* base, int ld, int kk) {
  int lane = threadIdx.x & 31;
  int half = lane >> 4;
  int r    = lane & 15;
  const bf16_t* p = base + (size_t)r * ld + kk + half * 8;
  v8bf lo = *(const v8bf*)p;
  v8bf hi = *(const v8bf*)(p + 16);
  v16bf out;
#pragma unroll
  for (int i = 0; i < 8; ++i) { out[i] = lo[i]; out[i + 8] = hi[i]; }
  return out;
}

// B fragment: 32x16 bf16. Source stored per-column-contiguous-in-K with
// column stride `ld`: element (k,c) at base[c*ld + k].
// lane<16: col=lane, K=kk+[0..8) and kk+[16..24); lane>=16: col=lane-16, K=kk+[8..16) and kk+[24..32)
__device__ inline v16bf load_b_frag(const bf16_t* base, int ld, int kk) {
  int lane = threadIdx.x & 31;
  int half = lane >> 4;
  int c    = lane & 15;
  const bf16_t* p = base + (size_t)c * ld + kk + half * 8;
  v8bf lo = *(const v8bf*)p;
  v8bf hi = *(const v8bf*)(p + 16);
  v16bf out;
#pragma unroll
  for (int i = 0; i < 8; ++i) { out[i] = lo[i]; out[i + 8] = hi[i]; }
  return out;
}

// ---------------- small reductions ----------------

__device__ inline float warp_sum(float v) {
#pragma unroll
  for (int off = 16; off > 0; off >>= 1) v += __shfl_xor(v, off, 32);
  return v;
}

// block of 64 threads
__device__ inline float blk64_sum(float v, float* sh2) {
  v = warp_sum(v);
  int w = threadIdx.x >> 5;
  if ((threadIdx.x & 31) == 0) sh2[w] = v;
  __syncthreads();
  float tot = sh2[0] + sh2[1];
  __syncthreads();
  return tot;
}

// ---------------- kernel: weight prep (Wk/Wv -> bf16, [DS][DIN]) ----------------

__global__ void prep_weights(const float* __restrict__ Wk, const float* __restrict__ Wv,
                             bf16_t* __restrict__ WkT, bf16_t* __restrict__ WvT) {
  int i = blockIdx.x * blockDim.x + threadIdx.x;   // 0 .. 2*16384
  int sel = i >> 14;
  int j   = i & 16383;
  int din = j >> 6;       // 0..255
  int d   = j & 63;       // 0..63
  const float* src = sel ? Wv : Wk;
  bf16_t* dst = sel ? WvT : WkT;
  dst[(size_t)d * DIN + din] = (bf16_t)src[(size_t)din * DSL + d];
}

// ---------------- kernel: slots init ----------------

__global__ void init_slots(const float* __restrict__ noise, const float* __restrict__ mu,
                           const float* __restrict__ logsig, float* __restrict__ slots) {
  int i = blockIdx.x * blockDim.x + threadIdx.x;   // 0 .. 64*20*64
  int d = i & 63;
  slots[i] = mu[d] + __expf(logsig[d]) * noise[i];
}

// ---------------- kernel: fused LN(inputs) -> k (row-major bf16), v^T (bf16) ----------------
// grid: (B*N)/64 blocks, 256 threads (8 waves). Each block: 64 rows.

__global__ void ln_kv_kernel(const float* __restrict__ x,
                             const float* __restrict__ g, const float* __restrict__ bta,
                             const bf16_t* __restrict__ WkT, const bf16_t* __restrict__ WvT,
                             bf16_t* __restrict__ kout, bf16_t* __restrict__ vT) {
  __shared__ bf16_t xs[64 * DIN];  // 32 KB
  const int wave = threadIdx.x >> 5;
  const int lane = threadIdx.x & 31;
  const size_t blockRow0 = (size_t)blockIdx.x * 64;
  const int b  = blockIdx.x >> 6;          // 64 blocks per batch
  const int n0 = (blockIdx.x & 63) * 64;

  // ---- stage 1: LayerNorm 64 rows into LDS as bf16
#pragma unroll
  for (int rr = 0; rr < 8; ++rr) {
    int row = wave * 8 + rr;
    const float* xr = x + (blockRow0 + row) * DIN + lane * 8;
    float v[8];
    float s = 0.f, ss = 0.f;
#pragma unroll
    for (int i = 0; i < 8; ++i) { v[i] = xr[i]; s += v[i]; ss += v[i] * v[i]; }
    s  = warp_sum(s);
    ss = warp_sum(ss);
    float mean = s * (1.0f / DIN);
    float var  = ss * (1.0f / DIN) - mean * mean;
    float inv  = rsqrtf(var + EPS_LN);
    bf16_t* dst = xs + row * DIN + lane * 8;
#pragma unroll
    for (int i = 0; i < 8; ++i) {
      int c = lane * 8 + i;
      dst[i] = (bf16_t)((v[i] - mean) * inv * g[c] + bta[c]);
    }
  }
  __syncthreads();

  // ---- stage 2: WMMA 16x64 tile per wave; waves 0-3 -> k, 4-7 -> v
  const int mat = wave >> 2;          // 0=k, 1=v
  const int rt  = wave & 3;           // row tile 0..3
  const bf16_t* WT = mat ? WvT : WkT;
  v8f acc[4] = {};
#pragma unroll
  for (int ks = 0; ks < 8; ++ks) {
    int kk = ks * 32;
    v16bf a = load_a_frag(xs + rt * 16 * DIN, DIN, kk);
#pragma unroll
    for (int ct = 0; ct < 4; ++ct) {
      v16bf bf = load_b_frag(WT + (size_t)(ct * 16) * DIN, DIN, kk);
      acc[ct] = wmma_bf16(a, bf, acc[ct]);
    }
  }
  const int half = lane >> 4;
  const int l15  = lane & 15;
  if (mat == 0) {
    // k row-major [row][d]
#pragma unroll
    for (int ct = 0; ct < 4; ++ct) {
#pragma unroll
      for (int r = 0; r < 8; ++r) {
        size_t row = blockRow0 + rt * 16 + r + 8 * half;
        kout[row * DSL + ct * 16 + l15] = (bf16_t)acc[ct][r];
      }
    }
  } else {
    // v transposed [b][d][n]; each lane holds 8 consecutive n for one d
#pragma unroll
    for (int ct = 0; ct < 4; ++ct) {
      int d = ct * 16 + l15;
      int n = n0 + rt * 16 + 8 * half;
      v8bf pack;
#pragma unroll
      for (int r = 0; r < 8; ++r) pack[r] = (bf16_t)acc[ct][r];
      *(v8bf*)(vT + (size_t)b * DSL * NPTS + (size_t)d * NPTS + n) = pack;
    }
  }
}

// ---------------- kernel: q = LN(slots) @ Wq, bf16, padded to 32 rows ----------------
// grid: B*KPAD blocks, 64 threads

__global__ void q_kernel(const float* __restrict__ slots,
                         const float* __restrict__ lng, const float* __restrict__ lnb,
                         const float* __restrict__ Wq, bf16_t* __restrict__ qbf) {
  int b = blockIdx.x >> 5;
  int m = blockIdx.x & 31;
  int t = threadIdx.x;
  if (m >= KSLOT) { qbf[((size_t)b * KPAD + m) * DSL + t] = (bf16_t)0.0f; return; }
  __shared__ float sh2[2];
  __shared__ float lnbuf[DSL];
  float v = slots[((size_t)b * KSLOT + m) * DSL + t];
  float mean = blk64_sum(v, sh2) * (1.0f / DSL);
  float d0 = v - mean;
  float var = blk64_sum(d0 * d0, sh2) * (1.0f / DSL);
  float inv = rsqrtf(var + EPS_LN);
  lnbuf[t] = d0 * inv * lng[t] + lnb[t];
  __syncthreads();
  float q = 0.f;
#pragma unroll 8
  for (int j = 0; j < DSL; ++j) q += lnbuf[j] * Wq[j * DSL + t];
  qbf[((size_t)b * KPAD + m) * DSL + t] = (bf16_t)q;
}

// ---------------- kernel: attention chunk ----------------
// grid: B*NCHUNK blocks, 256 threads (8 waves). NT=256 columns per block.
// dots = q@k^T*scale -> softmax over slots -> attn (bf16 in LDS) ->
// U_part = attn @ v^T-chunk, S_part = row sums.

#define PD (NT + 4)   // f32 dots row stride
#define PA NT         // bf16 attn row stride (512B, 16B-aligned chunks)

__global__ void attn_kernel(const bf16_t* __restrict__ qbf, const bf16_t* __restrict__ kbf,
                            const bf16_t* __restrict__ vT,
                            float* __restrict__ Upart, float* __restrict__ Spart,
                            float* __restrict__ attn_out, int write_attn) {
  __shared__ bf16_t qs[KPAD * DSL];       // 4 KB
  __shared__ float  dts[KPAD * PD];       // 33.3 KB
  __shared__ bf16_t ats[KPAD * PA];       // 16 KB
  const int chunk = blockIdx.x & (NCHUNK - 1);
  const int b     = blockIdx.x >> 4;
  const int wave  = threadIdx.x >> 5;
  const int lane  = threadIdx.x & 31;
  const int half  = lane >> 4;
  const int l15   = lane & 15;
  const int t     = threadIdx.x;

  // stage 0: stage q into LDS
#pragma unroll
  for (int i = 0; i < 8; ++i)
    qs[t + i * 256] = qbf[(size_t)b * KPAD * DSL + t + i * 256];
  __syncthreads();

  // stage 1: dots = q @ k^T (32 x 256), 32 tiles over 8 waves
  const bf16_t* kbase = kbf + ((size_t)b * NPTS + chunk * NT) * DSL;
#pragma unroll
  for (int p = 0; p < 4; ++p) {
    int pid = wave * 4 + p;
    int mt = pid & 1;
    int nt = pid >> 1;
    v8f acc = {};
#pragma unroll
    for (int ks = 0; ks < 2; ++ks) {
      int kk = ks * 32;
      v16bf a  = load_a_frag(qs + mt * 16 * DSL, DSL, kk);
      v16bf bf = load_b_frag(kbase + (size_t)(nt * 16) * DSL, DSL, kk);
      acc = wmma_bf16(a, bf, acc);
    }
#pragma unroll
    for (int r = 0; r < 8; ++r) {
      int m = mt * 16 + r + 8 * half;
      dts[m * PD + nt * 16 + l15] = acc[r] * SCALE;
    }
  }
  __syncthreads();

  // stage 2: softmax over the 20 slots for this thread's column
  {
    float d[KSLOT];
    float mx = -3.0e38f;
#pragma unroll
    for (int m = 0; m < KSLOT; ++m) { d[m] = dts[m * PD + t]; mx = fmaxf(mx, d[m]); }
    float s = 0.f;
#pragma unroll
    for (int m = 0; m < KSLOT; ++m) { d[m] = __expf(d[m] - mx); s += d[m]; }
    float rs = 1.0f / s;
#pragma unroll
    for (int m = 0; m < KSLOT; ++m) {
      float a = d[m] * rs;
      dts[m * PD + t] = a;                 // f32 copy for row sums
      ats[m * PA + t] = (bf16_t)a;         // bf16 A-operand
      if (write_attn)
        attn_out[((size_t)b * KSLOT + m) * NPTS + chunk * NT + t] = a;
    }
#pragma unroll
    for (int m = KSLOT; m < KPAD; ++m) ats[m * PA + t] = (bf16_t)0.0f;
  }
  __syncthreads();

  // stage 3: per-chunk attention row sums (20 rows)
  if (t < KSLOT) {
    float s = 0.f;
    for (int n = 0; n < NT; ++n) s += dts[t * PD + n];
    Spart[((size_t)b * NCHUNK + chunk) * KSLOT + t] = s;
  }
  __syncthreads();

  // stage 4: U_part = attn(32xNT) @ vT-chunk(NTx64); 8 tiles, one per wave
  {
    int mt = wave & 1;
    int ct = wave >> 1;
    const bf16_t* vbase = vT + (size_t)b * DSL * NPTS + (size_t)(ct * 16) * NPTS + chunk * NT;
    v8f acc = {};
#pragma unroll
    for (int ks = 0; ks < 8; ++ks) {
      int kk = ks * 32;
      v16bf a  = load_a_frag(ats + mt * 16 * PA, PA, kk);
      v16bf bf = load_b_frag(vbase, NPTS, kk);
      acc = wmma_bf16(a, bf, acc);
    }
#pragma unroll
    for (int r = 0; r < 8; ++r) {
      int m = mt * 16 + r + 8 * half;
      if (m < KSLOT)
        Upart[(((size_t)b * NCHUNK + chunk) * KSLOT + m) * DSL + ct * 16 + l15] = acc[r];
    }
  }
}

// ---------------- kernel: reduce + GRU + LN + MLP ----------------
// grid: B*KSLOT blocks, 64 threads

__global__ void update_kernel(const float* __restrict__ Upart, const float* __restrict__ Spart,
                              float* __restrict__ slots,
                              const float* __restrict__ Wih, const float* __restrict__ Whh,
                              const float* __restrict__ bih, const float* __restrict__ bhh,
                              const float* __restrict__ lng, const float* __restrict__ lnb,
                              const float* __restrict__ W1, const float* __restrict__ b1,
                              const float* __restrict__ W2, const float* __restrict__ b2,
                              float* __restrict__ out_slots, int write_out) {
  int t = threadIdx.x;
  int b = blockIdx.x / KSLOT;
  int m = blockIdx.x % KSLOT;
  __shared__ float sh2[2];
  __shared__ float updL[DSL], spL[DSL], hL[DSL], hidL[4 * DSL];

  float u = 0.f, s = 0.f;
#pragma unroll
  for (int ch = 0; ch < NCHUNK; ++ch) {
    u += Upart[(((size_t)b * NCHUNK + ch) * KSLOT + m) * DSL + t];
    s += Spart[((size_t)b * NCHUNK + ch) * KSLOT + m];
  }
  float upd = u / (s + EPS_A);
  float sp = slots[((size_t)b * KSLOT + m) * DSL + t];
  updL[t] = upd;
  spL[t] = sp;
  __syncthreads();

  float gi[3], gh[3];
#pragma unroll
  for (int gidx = 0; gidx < 3; ++gidx) {
    float ai = bih[gidx * DSL + t];
    float ah = bhh[gidx * DSL + t];
    const float* wi = Wih + (size_t)(gidx * DSL + t) * DSL;
    const float* wh = Whh + (size_t)(gidx * DSL + t) * DSL;
#pragma unroll 8
    for (int j = 0; j < DSL; ++j) { ai += updL[j] * wi[j]; ah += spL[j] * wh[j]; }
    gi[gidx] = ai; gh[gidx] = ah;
  }
  float r = 1.0f / (1.0f + __expf(-(gi[0] + gh[0])));
  float z = 1.0f / (1.0f + __expf(-(gi[1] + gh[1])));
  float nn = tanhf(gi[2] + r * gh[2]);
  float snew = (1.0f - z) * nn + z * sp;

  float mean = blk64_sum(snew, sh2) * (1.0f / DSL);
  float d0 = snew - mean;
  float var = blk64_sum(d0 * d0, sh2) * (1.0f / DSL);
  float h = d0 * rsqrtf(var + EPS_LN) * lng[t] + lnb[t];
  hL[t] = h;
  __syncthreads();

#pragma unroll
  for (int i4 = 0; i4 < 4; ++i4) {
    int idx = i4 * DSL + t;
    float a = b1[idx];
#pragma unroll 8
    for (int j = 0; j < DSL; ++j) a += hL[j] * W1[(size_t)j * 256 + idx];
    hidL[idx] = fmaxf(a, 0.0f);
  }
  __syncthreads();

  float o = b2[t];
#pragma unroll 8
  for (int i = 0; i < 256; ++i) o += hidL[i] * W2[(size_t)i * DSL + t];
  float sfinal = snew + o;
  slots[((size_t)b * KSLOT + m) * DSL + t] = sfinal;
  if (write_out) out_slots[((size_t)b * KSLOT + m) * DSL + t] = sfinal;
}

// ---------------- host launch ----------------

extern "C" void kernel_launch(void* const* d_in, const int* in_sizes, int n_in,
                              void* d_out, int out_size, void* d_ws, size_t ws_size,
                              hipStream_t stream) {
  (void)in_sizes; (void)n_in; (void)out_size; (void)ws_size;
  const float* inputs    = (const float*)d_in[0];
  const float* noise     = (const float*)d_in[1];
  const float* mu        = (const float*)d_in[2];
  const float* logsig    = (const float*)d_in[3];
  const float* ln_in_g   = (const float*)d_in[4];
  const float* ln_in_b   = (const float*)d_in[5];
  const float* Wk        = (const float*)d_in[6];
  const float* Wv        = (const float*)d_in[7];
  const float* Wq        = (const float*)d_in[8];
  const float* ln_slot_g = (const float*)d_in[9];
  const float* ln_slot_b = (const float*)d_in[10];
  const float* gru_Wih   = (const float*)d_in[11];
  const float* gru_Whh   = (const float*)d_in[12];
  const float* gru_bih   = (const float*)d_in[13];
  const float* gru_bhh   = (const float*)d_in[14];
  const float* ln_mlp_g  = (const float*)d_in[15];
  const float* ln_mlp_b  = (const float*)d_in[16];
  const float* mlp_W1    = (const float*)d_in[17];
  const float* mlp_b1    = (const float*)d_in[18];
  const float* mlp_W2    = (const float*)d_in[19];
  const float* mlp_b2    = (const float*)d_in[20];

  char* ws = (char*)d_ws;
  bf16_t* kbf   = (bf16_t*)(ws);                       // 32 MB
  bf16_t* vT    = (bf16_t*)(ws + 33554432);            // 32 MB
  bf16_t* WkT   = (bf16_t*)(ws + 67108864);            // 32 KB
  bf16_t* WvT   = (bf16_t*)(ws + 67141632);            // 32 KB
  bf16_t* qbf   = (bf16_t*)(ws + 67174400);            // 256 KB
  float*  slots = (float*) (ws + 67436544);            // 320 KB
  float*  Upart = (float*) (ws + 67764224);            // 5 MB
  float*  Spart = (float*) (ws + 73007104);            // 80 KB

  float* out_slots = (float*)d_out;
  float* out_attn  = (float*)d_out + (size_t)BATCH * KSLOT * DSL;

  prep_weights<<<128, 256, 0, stream>>>(Wk, Wv, WkT, WvT);
  init_slots<<<(BATCH * KSLOT * DSL) / 256, 256, 0, stream>>>(noise, mu, logsig, slots);
  ln_kv_kernel<<<(BATCH * NPTS) / 64, 256, 0, stream>>>(inputs, ln_in_g, ln_in_b,
                                                        WkT, WvT, kbf, vT);
  for (int it = 0; it < 3; ++it) {
    int last = (it == 2);
    q_kernel<<<BATCH * KPAD, 64, 0, stream>>>(slots, ln_slot_g, ln_slot_b, Wq, qbf);
    attn_kernel<<<BATCH * NCHUNK, 256, 0, stream>>>(qbf, kbf, vT, Upart, Spart,
                                                    out_attn, last);
    update_kernel<<<BATCH * KSLOT, 64, 0, stream>>>(Upart, Spart, slots,
                                                    gru_Wih, gru_Whh, gru_bih, gru_bhh,
                                                    ln_mlp_g, ln_mlp_b, mlp_W1, mlp_b1,
                                                    mlp_W2, mlp_b2, out_slots, last);
  }
}
